// KBertGATEnricher_1331439862468
// MI455X (gfx1250) — matrix operand order repaired and checked by reading, not compiled, not run on hardware
//
#include <hip/hip_runtime.h>
#include <math.h>

// ---------------- problem constants ----------------
#define BATCH   8
#define SEQ     256
#define BN      2048          // BATCH*SEQ
#define HID     768
#define NHEADS  4
#define FDIM    128
#define VOCAB   30522
#define KCAT    512           // NHEADS*FDIM
#define VTILES  239           // ceil(VOCAB/128)
#define LN_EPS  1e-12f
#define LRELU_A 0.01f
#define NEGBIG  (-9.0e15f)

// ---------------- workspace layout (bytes) ----------------
#define MASK_OFF 0u                    // u8  [B][N][N]              524288
#define X32_OFF  524288u               // f32 [BN][HID]              6291456
#define XBF_OFF  6815744u              // bf16[BN][HID]              3145728
#define WT_OFF   9961472u              // bf16[H][F][HID]  (W^T)     786432
#define WH_OFF   10747904u             // f32 [H][BN][F]             4194304
#define WHT_OFF  14942208u             // bf16[H*B][F][N]  (Wh^T)    2097152
#define E1_OFF   17039360u             // f32 [H][BN]                32768
#define E2_OFF   17072128u             // f32 [H][BN]                32768
#define ATT_OFF  17104896u             // bf16[H*B][N][N]            4194304
#define CAT_OFF  21299200u             // bf16[BN][KCAT]             2097152
#define OWB_OFF  23396352u             // bf16[VOCAB][KCAT]          31254528
#define PT_OFF   54650880u             // f32 [BN][VTILES][2]        3915776
#define LSE_OFF  58566912u             // f32 [BN]                   8192
// total ~58.6 MB

// ---------------- vector types ----------------
typedef __attribute__((ext_vector_type(16))) __bf16          v16bf;
typedef __attribute__((ext_vector_type(8)))  float           v8f;
typedef __attribute__((ext_vector_type(8)))  unsigned short  ush8;
typedef __attribute__((ext_vector_type(16))) unsigned short  ush16;

__device__ __forceinline__ unsigned short f2bf(float f) {
  unsigned int u = __builtin_bit_cast(unsigned int, f);
  unsigned int r = u + 0x7FFFu + ((u >> 16) & 1u);
  return (unsigned short)(r >> 16);
}

// A-matrix fragment (16x32 bf16): lane l<16 -> M=l, K={0..7,16..23};
// lane l>=16 -> M=l-16, K={8..15,24..31}.  Two 16B chunks per lane.
__device__ __forceinline__ v16bf load_fragA(const unsigned short* base, int ld,
                                            int row0, int k0, int lane) {
  int r  = row0 + (lane & 15);
  int kb = k0 + ((lane >> 4) << 3);
  const ush8* p0 = reinterpret_cast<const ush8*>(base + (size_t)r * ld + kb);
  const ush8* p1 = reinterpret_cast<const ush8*>(base + (size_t)r * ld + kb + 16);
  ush16 t = __builtin_shufflevector(*p0, *p1,
      0,1,2,3,4,5,6,7,8,9,10,11,12,13,14,15);
  return __builtin_bit_cast(v16bf, t);
}

// B-matrix fragment (32x16 bf16) from Bt (output-col-major, K contiguous):
// lane l<16 -> N=l, K={0..7,16..23} ; lane l>=16 -> N=l-16, K={8..15,24..31}.
__device__ __forceinline__ v16bf load_fragB(const unsigned short* baseT, int ld,
                                            int col0, int k0, int lane) {
  int c  = col0 + (lane & 15);
  int kb = k0 + ((lane >> 4) << 3);
  const ush8* p0 = reinterpret_cast<const ush8*>(baseT + (size_t)c * ld + kb);
  const ush8* p1 = reinterpret_cast<const ush8*>(baseT + (size_t)c * ld + kb + 16);
  ush16 t = __builtin_shufflevector(*p0, *p1,
      0,1,2,3,4,5,6,7,8,9,10,11,12,13,14,15);
  return __builtin_bit_cast(v16bf, t);
}

// One wave computes a 16(M) x 128(N) strip: 8 wmma tiles, A reused 8x.
// (No software prefetch: the compiler pipelines the clause'd b128 loads
//  against the WMMAs at s_wait_loadcnt 0x1 on its own; an explicit
//  __builtin_prefetch was verified to break that pipelining.)
__device__ __forceinline__ void wave_gemm_16x128(
    const unsigned short* A, int lda, int arow0,
    const unsigned short* Bt, int ldb, int bcol0, int colMax,
    int K, int lane, v8f acc[8]) {
  for (int k0 = 0; k0 < K; k0 += 32) {
    v16bf a = load_fragA(A, lda, arow0, k0, lane);
#pragma unroll
    for (int t = 0; t < 8; ++t) {
      int c0 = bcol0 + t * 16;
      v16bf b = {};
      if (c0 + (lane & 15) < colMax) b = load_fragB(Bt, ldb, c0, k0, lane);
      acc[t] = __builtin_amdgcn_wmma_f32_16x16x32_bf16(
          false, a, false, b, (short)0, acc[t], false, false);
    }
  }
}

// ================= kernel 0: graph mask =================
__global__ void k_mask(const int* __restrict__ tok, const int* __restrict__ typ,
                       const int* __restrict__ syn, const int* __restrict__ hw,
                       unsigned char* __restrict__ mask) {
  int idx = blockIdx.x * blockDim.x + threadIdx.x;
  if (idx >= BATCH * SEQ * SEQ) return;
  int b = idx >> 16, rem = idx & 65535;
  int i = rem >> 8, j = rem & 255;
  int bi = b * SEQ + i, bj = b * SEQ + j;
  int ti = typ[bi], tj = typ[bj];
  bool hi = hw[bi] != 0, hj = hw[bj] != 0;
  bool vis = (syn[bi] == syn[bj]);
  bool s1 = (ti == 1) && hi, s3 = (ti == 3) && hi;
  bool d1 = ((tj == 0) || (tj == 2) || (tj == 5)) && hj;
  bool d3 = ((tj == 6) || (tj == 4) || (tj == 0)) && hj;
  vis = vis || (s1 && d1) || (s3 && d3);
  vis = vis && (tok[bj] != 0);
  mask[idx] = vis ? 1 : 0;
}

// ================= convert out_W to bf16 =================
__global__ void k_conv_outw(const float* __restrict__ src,
                            unsigned short* __restrict__ dst, int n) {
  int i = blockIdx.x * blockDim.x + threadIdx.x;
  if (i < n) dst[i] = f2bf(src[i]);
}

// ============ transpose W (H,HID,F) -> Wt bf16 (H,F,HID) ============
__global__ void k_conv_wt(const float* __restrict__ W,
                          unsigned short* __restrict__ Wt) {
  int i = blockIdx.x * blockDim.x + threadIdx.x;
  if (i >= NHEADS * HID * FDIM) return;
  int h = i / (HID * FDIM);
  int r = i - h * HID * FDIM;
  int d = r / FDIM, f = r - d * FDIM;
  Wt[((size_t)h * FDIM + f) * HID + d] = f2bf(W[i]);
}

// ============== kernel: embed + layernorm (one block per token) ==============
__global__ void k_embed_ln(const int* __restrict__ tok, const int* __restrict__ typ,
                           const float* __restrict__ te, const float* __restrict__ ye,
                           const float* __restrict__ pe, const float* __restrict__ g,
                           const float* __restrict__ be,
                           float* __restrict__ x32, unsigned short* __restrict__ xbf) {
  __shared__ float red[256];
  int m = blockIdx.x;              // token index in [0, BN)
  int n = m & (SEQ - 1);
  int tid = threadIdx.x;
  int t = tok[m], y = typ[m];
  float v[3];
  float s = 0.f;
#pragma unroll
  for (int q = 0; q < 3; ++q) {
    int d = tid + q * 256;
    v[q] = te[(size_t)t * HID + d] + ye[(size_t)y * HID + d] + pe[(size_t)n * HID + d];
    s += v[q];
  }
  red[tid] = s; __syncthreads();
  for (int st = 128; st > 0; st >>= 1) {
    if (tid < st) red[tid] += red[tid + st];
    __syncthreads();
  }
  float mu = red[0] / (float)HID;
  __syncthreads();
  s = 0.f;
#pragma unroll
  for (int q = 0; q < 3; ++q) { float d = v[q] - mu; s += d * d; }
  red[tid] = s; __syncthreads();
  for (int st = 128; st > 0; st >>= 1) {
    if (tid < st) red[tid] += red[tid + st];
    __syncthreads();
  }
  float rs = rsqrtf(red[0] / (float)HID + LN_EPS);
#pragma unroll
  for (int q = 0; q < 3; ++q) {
    int d = tid + q * 256;
    float yv = (v[q] - mu) * rs * g[d] + be[d];
    x32[(size_t)m * HID + d] = yv;
    xbf[(size_t)m * HID + d] = f2bf(yv);
  }
}

// ============== kernel: Wh = x @ W[h]  (WMMA bf16) ==============
__global__ void k_gemm_wh(const unsigned short* __restrict__ xbf,
                          const unsigned short* __restrict__ wt,
                          float* __restrict__ wh,
                          unsigned short* __restrict__ whT) {
  int h = blockIdx.y;
  int wave = threadIdx.x >> 5, lane = threadIdx.x & 31;
  int arow0 = blockIdx.x * 128 + wave * 16;
  v8f acc[8] = {};
  wave_gemm_16x128(xbf, HID, arow0, wt + (size_t)h * FDIM * HID, HID,
                   0, FDIM, HID, lane, acc);
  int sel = lane >> 4, col = lane & 15;
#pragma unroll
  for (int t = 0; t < 8; ++t) {
#pragma unroll
    for (int r = 0; r < 8; ++r) {
      int m = arow0 + r + 8 * sel;
      int n = t * 16 + col;
      float vv = acc[t][r];
      wh[((size_t)h * BN + m) * FDIM + n] = vv;
      int b = m >> 8, i = m & 255;
      whT[(((size_t)(h * BATCH + b)) * FDIM + n) * SEQ + i] = f2bf(vv);
    }
  }
}

// ============== kernel: e1,e2 dots ==============
__global__ void k_escore(const float* __restrict__ wh, const float* __restrict__ a,
                         float* __restrict__ e1, float* __restrict__ e2) {
  int gid = blockIdx.x * blockDim.x + threadIdx.x;
  if (gid >= NHEADS * BN) return;
  int h = gid >> 11, m = gid & (BN - 1);
  const float* row = wh + ((size_t)h * BN + m) * FDIM;
  const float* a1 = a + h * 2 * FDIM;
  const float* a2 = a1 + FDIM;
  float s1 = 0.f, s2 = 0.f;
#pragma unroll 4
  for (int f = 0; f < FDIM; ++f) { s1 += row[f] * a1[f]; s2 += row[f] * a2[f]; }
  e1[gid] = s1; e2[gid] = s2;
}

// ===== kernel: att = softmax over i (axis=2) of lrelu(e1_i+e2_j) masked =====
// one block per (h,b); thread j owns a column.
__global__ void k_att(const float* __restrict__ e1, const float* __restrict__ e2,
                      const unsigned char* __restrict__ mask,
                      unsigned short* __restrict__ att) {
  __shared__ float se1[SEQ];
  int hb = blockIdx.x;            // h*8 + b
  int h = hb >> 3, b = hb & 7;
  int j = threadIdx.x;
  se1[j] = e1[h * BN + b * SEQ + j];
  __syncthreads();
  float e2j = e2[h * BN + b * SEQ + j];
  const unsigned char* mcol = mask + (size_t)b * SEQ * SEQ + j;
  float mx = -INFINITY, sm = 0.f;
  for (int i = 0; i < SEQ; ++i) {
    float v = se1[i] + e2j;
    v = v > 0.f ? v : LRELU_A * v;
    if (!mcol[(size_t)i * SEQ]) v = NEGBIG;
    if (v > mx) { sm = sm * __expf(mx - v) + 1.f; mx = v; }
    else        { sm += __expf(v - mx); }
  }
  float inv = 1.f / sm;
  unsigned short* acol = att + (size_t)hb * SEQ * SEQ + j;
  for (int i = 0; i < SEQ; ++i) {
    float v = se1[i] + e2j;
    v = v > 0.f ? v : LRELU_A * v;
    if (!mcol[(size_t)i * SEQ]) v = NEGBIG;
    acol[(size_t)i * SEQ] = f2bf(__expf(v - mx) * inv);
  }
}

// ============== kernel: hp = elu(att @ Wh) -> cat bf16 ==============
__global__ void k_gemm_hp(const unsigned short* __restrict__ att,
                          const unsigned short* __restrict__ whT,
                          unsigned short* __restrict__ cat) {
  int hb = blockIdx.y;
  int h = hb >> 3, b = hb & 7;
  int wave = threadIdx.x >> 5, lane = threadIdx.x & 31;
  int arow0 = blockIdx.x * 128 + wave * 16;
  v8f acc[8] = {};
  wave_gemm_16x128(att + (size_t)hb * SEQ * SEQ, SEQ, arow0,
                   whT + (size_t)hb * FDIM * SEQ, SEQ,
                   0, FDIM, SEQ, lane, acc);
  int sel = lane >> 4, col = lane & 15;
#pragma unroll
  for (int t = 0; t < 8; ++t) {
#pragma unroll
    for (int r = 0; r < 8; ++r) {
      int i = arow0 + r + 8 * sel;
      int f = t * 16 + col;
      float vv = acc[t][r];
      vv = vv > 0.f ? vv : (__expf(vv) - 1.f);        // elu
      cat[((size_t)(b * SEQ + i)) * KCAT + h * FDIM + f] = f2bf(vv);
    }
  }
}

// ==== kernel: out = elu(cat @ out_W^T + b) + per-tile logsumexp partials ====
__global__ void k_gemm_out(const unsigned short* __restrict__ cat,
                           const unsigned short* __restrict__ owb,
                           const float* __restrict__ bias,
                           float* __restrict__ out,
                           float* __restrict__ ptile) {
  int wave = threadIdx.x >> 5, lane = threadIdx.x & 31;
  int v0 = blockIdx.x * 128;
  int arow0 = blockIdx.y * 128 + wave * 16;
  v8f acc[8] = {};
  wave_gemm_16x128(cat, KCAT, arow0, owb, KCAT, v0, VOCAB, KCAT, lane, acc);
  int sel = lane >> 4, col = lane & 15;
  float bv[8];
#pragma unroll
  for (int t = 0; t < 8; ++t) {
    int v = v0 + t * 16 + col;
    bv[t] = (v < VOCAB) ? bias[v] : 0.f;
  }
#pragma unroll
  for (int r = 0; r < 8; ++r) {
    int m = arow0 + r + 8 * sel;
    float mx = -INFINITY, sm = 0.f;
#pragma unroll
    for (int t = 0; t < 8; ++t) {
      int v = v0 + t * 16 + col;
      if (v < VOCAB) {
        float vv = acc[t][r] + bv[t];
        vv = vv > 0.f ? vv : (__expf(vv) - 1.f);      // elu
        out[(size_t)m * VOCAB + v] = vv;
        if (vv > mx) { sm = sm * __expf(mx - vv) + 1.f; mx = vv; }
        else         { sm += __expf(vv - mx); }
      }
    }
    // reduce (mx,sm) across the 16 lanes of this half-wave (masks <= 8)
#pragma unroll
    for (int off = 1; off <= 8; off <<= 1) {
      float m2 = __shfl_xor(mx, off, 32);
      float s2 = __shfl_xor(sm, off, 32);
      if (m2 > mx)                    { sm = sm * __expf(mx - m2) + s2; mx = m2; }
      else if (m2 != -INFINITY)       { sm += s2 * __expf(m2 - mx); }
    }
    if (col == 0) {
      size_t p = ((size_t)m * VTILES + blockIdx.x) * 2;
      ptile[p]     = mx;
      ptile[p + 1] = sm;
    }
  }
}

// ============== kernel: combine per-tile partials -> lse[m] ==============
__global__ void k_lse(const float* __restrict__ ptile, float* __restrict__ lse) {
  int m = blockIdx.x * blockDim.x + threadIdx.x;
  if (m >= BN) return;
  const float* p = ptile + (size_t)m * VTILES * 2;
  float mx = -INFINITY, sm = 0.f;
  for (int t = 0; t < VTILES; ++t) {
    float pm = p[2 * t], ps = p[2 * t + 1];
    if (pm > mx)       { sm = sm * __expf(mx - pm) + ps; mx = pm; }
    else if (ps > 0.f) { sm += ps * __expf(pm - mx); }
  }
  lse[m] = mx + __logf(sm);
}

// ============== kernel: out -= lse[row]  (single read+write pass) ==============
__global__ void k_sub_lse(float* __restrict__ out, const float* __restrict__ lse) {
  int m = blockIdx.y;
  int c = blockIdx.x * blockDim.x + threadIdx.x;
  if (c < VOCAB) {
    size_t idx = (size_t)m * VOCAB + c;
    out[idx] = out[idx] - lse[m];
  }
}

// ================================================================
extern "C" void kernel_launch(void* const* d_in, const int* in_sizes, int n_in,
                              void* d_out, int out_size, void* d_ws, size_t ws_size,
                              hipStream_t stream) {
  const int*   tok   = (const int*)d_in[0];
  const int*   typ   = (const int*)d_in[1];
  const int*   syn   = (const int*)d_in[2];
  const int*   hw    = (const int*)d_in[3];
  const float* temb  = (const float*)d_in[4];
  const float* yemb  = (const float*)d_in[5];
  const float* pemb  = (const float*)d_in[6];
  const float* lng   = (const float*)d_in[7];
  const float* lnb   = (const float*)d_in[8];
  const float* W     = (const float*)d_in[9];
  const float* a     = (const float*)d_in[10];
  const float* outW  = (const float*)d_in[11];
  const float* outb  = (const float*)d_in[12];
  float* out = (float*)d_out;

  char* ws = (char*)d_ws;
  unsigned char*  mask  = (unsigned char*) (ws + MASK_OFF);
  float*          x32   = (float*)         (ws + X32_OFF);
  unsigned short* xbf   = (unsigned short*)(ws + XBF_OFF);
  unsigned short* wt    = (unsigned short*)(ws + WT_OFF);
  float*          wh    = (float*)         (ws + WH_OFF);
  unsigned short* whT   = (unsigned short*)(ws + WHT_OFF);
  float*          e1    = (float*)         (ws + E1_OFF);
  float*          e2    = (float*)         (ws + E2_OFF);
  unsigned short* att   = (unsigned short*)(ws + ATT_OFF);
  unsigned short* cat   = (unsigned short*)(ws + CAT_OFF);
  unsigned short* owb   = (unsigned short*)(ws + OWB_OFF);
  float*          ptile = (float*)         (ws + PT_OFF);
  float*          lse   = (float*)         (ws + LSE_OFF);

  // 0) graph mask
  k_mask<<<(BATCH * SEQ * SEQ) / 256, 256, 0, stream>>>(tok, typ, syn, hw, mask);
  // 1) weight conversions
  k_conv_outw<<<(VOCAB * KCAT + 255) / 256, 256, 0, stream>>>(outW, owb, VOCAB * KCAT);
  k_conv_wt<<<(NHEADS * HID * FDIM + 255) / 256, 256, 0, stream>>>(W, wt);
  // 2) embed + layernorm
  k_embed_ln<<<BN, 256, 0, stream>>>(tok, typ, temb, yemb, pemb, lng, lnb, x32, xbf);
  // 3) Wh = x @ W[h]  (WMMA)
  k_gemm_wh<<<dim3(BN / 128, NHEADS), 256, 0, stream>>>(xbf, wt, wh, whT);
  // 4) attention logits components
  k_escore<<<(NHEADS * BN) / 256, 256, 0, stream>>>(wh, a, e1, e2);
  // 5) masked column softmax -> att (bf16)
  k_att<<<NHEADS * BATCH, SEQ, 0, stream>>>(e1, e2, mask, att);
  // 6) hp = elu(att @ Wh) -> cat (WMMA)
  k_gemm_hp<<<dim3(SEQ / 128, NHEADS * BATCH), 256, 0, stream>>>(att, whT, cat);
  // 7) out = elu(cat @ out_W^T + b) + logsumexp partials (WMMA, dominant GEMM)
  k_gemm_out<<<dim3(VTILES, BN / 128), 256, 0, stream>>>(cat, owb, outb, out, ptile);
  // 8) combine partials -> lse, then one read+write normalization pass
  k_lse<<<BN / 256, 256, 0, stream>>>(ptile, lse);
  k_sub_lse<<<dim3((VOCAB + 255) / 256, BN), 256, 0, stream>>>(out, lse);
}
